// Ernie4_5Attention_91044716741344
// MI455X (gfx1250) — matrix operand order repaired and checked by reading, not compiled, not run on hardware
//
#include <hip/hip_runtime.h>
#include <hip/hip_bf16.h>

// ---------------------------------------------------------------------------
// Ernie4.5 attention for MI455X (gfx1250, wave32, WMMA).
// Pipeline: [GEMM qkv (bf16 wmma)] -> [rope] -> [flash attn (bf16 wmma)]
//           -> [GEMM out proj (bf16 wmma, f32 out)]
// Shapes: B=1, S=2048, HID=4096, H=32, KVH=8, D=128.
// ---------------------------------------------------------------------------

typedef __attribute__((ext_vector_type(16))) __bf16 v16bf;
typedef __attribute__((ext_vector_type(8)))  float  v8f;

#define S_LEN 2048
#define HID   4096
#define NH    32
#define NKVH  8
#define HD    128

// Optional CDNA5 async global->LDS staging (ASYNCcnt path), guarded so the
// kernel still compiles (with plain b128 copies) if the builtins are absent.
#if defined(__has_builtin)
# if __has_builtin(__builtin_amdgcn_global_load_async_to_lds_b128) && \
     __has_builtin(__builtin_amdgcn_s_wait_asynccnt)
#  define USE_ASYNC_LDS 1
# endif
#endif
#ifndef USE_ASYNC_LDS
# define USE_ASYNC_LDS 0
#endif

#if USE_ASYNC_LDS
typedef int v4i_t __attribute__((ext_vector_type(4)));
typedef __attribute__((address_space(1))) v4i_t* as1_v4i_p;   // global
typedef __attribute__((address_space(3))) v4i_t* as3_v4i_p;   // LDS
#endif

__device__ __forceinline__ unsigned short f2bf(float f) {
    union { float f; unsigned u; } v; v.f = f;
    unsigned r = v.u + 0x7FFFu + ((v.u >> 16) & 1u);   // round-nearest-even
    return (unsigned short)(r >> 16);
}
__device__ __forceinline__ unsigned pack2bf(float lo, float hi) {
    return (unsigned)f2bf(lo) | ((unsigned)f2bf(hi) << 16);
}
__device__ __forceinline__ float bf2f(unsigned short h) {
    union { unsigned u; float f; } v; v.u = ((unsigned)h) << 16;
    return v.f;
}

// A-fragment (16x32 bf16): lane holds row m = lane%16.
// element e -> k = (e/8)*16 + (lane/16)*8 + (e%8)  => two contiguous 16B reads.
__device__ __forceinline__ v16bf load_fragA(const unsigned short* rowBase, int half) {
    union { unsigned u[8]; v16bf v; } f;
    const unsigned* p0 = (const unsigned*)(rowBase + half * 8);
    const unsigned* p1 = (const unsigned*)(rowBase + 16 + half * 8);
#pragma unroll
    for (int j = 0; j < 4; ++j) { f.u[j] = p0[j]; f.u[4 + j] = p1[j]; }
    return f.v;
}
// B-fragment (32x16 bf16): lane holds col n = lane%16.
// element e -> k = (e/8)*16 + (lane/16)*8 + (e%8)  => two contiguous 16B reads.
__device__ __forceinline__ v16bf load_fragB(const unsigned short* rowBase, int half) {
    union { unsigned u[8]; v16bf v; } f;
    const unsigned* p0 = (const unsigned*)(rowBase + half * 8);
    const unsigned* p1 = (const unsigned*)(rowBase + 16 + half * 8);
#pragma unroll
    for (int j = 0; j < 4; ++j) { f.u[j] = p0[j]; f.u[4 + j] = p1[j]; }
    return f.v;
}

// ---------------------------------------------------------------------------
// GEMM: C[MxN] = A[MxK] * B[KxN].  A: f32 or bf16 row-major. B: f32 row-major.
// Block tile 128x64, BK=32, 256 threads (8 waves), wave tile 32x32 (4 WMMAs,
// 2 A-frags x 2 B-frags reused per stage).
// ---------------------------------------------------------------------------
#define BM 128
#define BN 64
#define BK 32

template <bool A_BF16, bool OUT_BF16>
__global__ __launch_bounds__(256) void gemm_wmma_kernel(
    const void* __restrict__ Av, const float* __restrict__ B,
    void* __restrict__ Cv, int M, int N, int K)
{
    __shared__ unsigned short sA[BM][BK + 8];   // row stride 40 (80B, 16B-aligned)
    __shared__ unsigned short sB[BN][BK + 8];   // stored transposed: [n][k]

    const int tid  = threadIdx.x;
    const int lane = tid & 31;
    const int wid  = tid >> 5;
    const int wm   = wid & 3;      // 4 wave rows of 32
    const int wn   = wid >> 2;     // 2 wave cols of 32
    const int bm   = blockIdx.y * BM;
    const int bn   = blockIdx.x * BN;
    const int half = lane >> 4;
    const int ln   = lane & 15;

    const float*          Af = (const float*)Av;
    const unsigned short* Ah = (const unsigned short*)Av;

    // staging coordinates: A 128x32 -> 16 elems/thread; B 32x64 -> 8 elems/thread
    const int arow = tid >> 1;
    const int acol = (tid & 1) * 16;
    const int brow = tid >> 2;          // n index
    const int bcol = (tid & 3) * 8;     // k index

    v8f acc00 = {0.f,0.f,0.f,0.f,0.f,0.f,0.f,0.f};
    v8f acc01 = acc00, acc10 = acc00, acc11 = acc00;

    for (int k0 = 0; k0 < K; k0 += BK) {
        {   // stage A tile (128x32): two 16B LDS stores per thread
            uint4* dst = (uint4*)&sA[arow][acol];
            if (A_BF16) {
                const unsigned short* src = Ah + (size_t)(bm + arow) * K + k0 + acol;
                if (k0 + BK < K) __builtin_prefetch(src + BK, 0, 1);
                dst[0] = ((const uint4*)src)[0];
                dst[1] = ((const uint4*)src)[1];
            } else {
                const float* src = Af + (size_t)(bm + arow) * K + k0 + acol;
                if (k0 + BK < K) __builtin_prefetch(src + BK, 0, 1);
#pragma unroll
                for (int j = 0; j < 2; ++j) {
                    float4 f0 = ((const float4*)src)[2 * j];
                    float4 f1 = ((const float4*)src)[2 * j + 1];
                    uint4 u;
                    u.x = pack2bf(f0.x, f0.y); u.y = pack2bf(f0.z, f0.w);
                    u.z = pack2bf(f1.x, f1.y); u.w = pack2bf(f1.z, f1.w);
                    dst[j] = u;
                }
            }
        }
        {   // stage B tile transposed -> sB[n][k] (strided reads, one 16B LDS store)
            const float* bsrc = B + (size_t)(k0 + bcol) * N + bn + brow;
            if (k0 + BK < K) __builtin_prefetch(bsrc + (size_t)BK * N, 0, 1);
            uint4 u;
            u.x = pack2bf(bsrc[0],            bsrc[(size_t)N]);
            u.y = pack2bf(bsrc[(size_t)2*N],  bsrc[(size_t)3*N]);
            u.z = pack2bf(bsrc[(size_t)4*N],  bsrc[(size_t)5*N]);
            u.w = pack2bf(bsrc[(size_t)6*N],  bsrc[(size_t)7*N]);
            *(uint4*)&sB[brow][bcol] = u;
        }
        __syncthreads();

        v16bf a0 = load_fragA(&sA[wm * 32 + ln][0], half);
        v16bf a1 = load_fragA(&sA[wm * 32 + 16 + ln][0], half);
        v16bf b0 = load_fragB(&sB[wn * 32 + ln][0], half);
        v16bf b1 = load_fragB(&sB[wn * 32 + 16 + ln][0], half);
        acc00 = __builtin_amdgcn_wmma_f32_16x16x32_bf16(false, a0, false, b0,
                                                        (short)0, acc00, false, false);
        acc01 = __builtin_amdgcn_wmma_f32_16x16x32_bf16(false, a0, false, b1,
                                                        (short)0, acc01, false, false);
        acc10 = __builtin_amdgcn_wmma_f32_16x16x32_bf16(false, a1, false, b0,
                                                        (short)0, acc10, false, false);
        acc11 = __builtin_amdgcn_wmma_f32_16x16x32_bf16(false, a1, false, b1,
                                                        (short)0, acc11, false, false);
        __syncthreads();
    }

    // C/D layout: element i -> row = (lane/16)*8 + i, col = lane%16
#pragma unroll
    for (int sub = 0; sub < 2; ++sub) {
        v8f aL = sub ? acc10 : acc00;
        v8f aR = sub ? acc11 : acc01;
#pragma unroll
        for (int i = 0; i < 8; ++i) {
            int row = bm + wm * 32 + sub * 16 + half * 8 + i;
            int c0  = bn + wn * 32 + ln;
            if (OUT_BF16) {
                unsigned short* Ch = (unsigned short*)Cv;
                Ch[(size_t)row * N + c0]      = f2bf(aL[i]);
                Ch[(size_t)row * N + c0 + 16] = f2bf(aR[i]);
            } else {
                float* Cf = (float*)Cv;
                Cf[(size_t)row * N + c0]      = aL[i];
                Cf[(size_t)row * N + c0 + 16] = aR[i];
            }
        }
    }
}

// ---------------------------------------------------------------------------
// RoPE (interleaved-pair convention): out[2i]=x0*c-x1*s, out[2i+1]=x1*c+x0*s
// One aligned dword per rotary pair. buf: [S, nheads*128] bf16, in place.
// ---------------------------------------------------------------------------
__global__ __launch_bounds__(256) void rope_kernel(unsigned short* buf, int nheads)
{
    int idx = blockIdx.x * blockDim.x + threadIdx.x;
    int total = S_LEN * nheads * (HD / 2);
    if (idx >= total) return;
    int i = idx % (HD / 2);
    int h = (idx / (HD / 2)) % nheads;
    int s = idx / ((HD / 2) * nheads);

    float freq = __powf(10000.0f, -(2.0f * (float)i) / (float)HD);
    float ang  = (float)s * freq;
    float sn, cs;
    __sincosf(ang, &sn, &cs);

    unsigned* p = (unsigned*)(buf + ((size_t)s * nheads + h) * HD + 2 * i);
    unsigned v = *p;
    float x0 = bf2f((unsigned short)(v & 0xffffu));
    float x1 = bf2f((unsigned short)(v >> 16));
    *p = pack2bf(x0 * cs - x1 * sn, x1 * cs + x0 * sn);
}

// ---------------------------------------------------------------------------
// Flash attention (causal, GQA 4:1). Block = 256 thr = 8 waves = 128 q rows.
// Each wave owns 16 q rows and the full 128-dim output accumulator.
// K chunk: 64 keys. Q fragments kept in registers across the whole kernel.
// ---------------------------------------------------------------------------
__global__ __launch_bounds__(256) void flash_attn_kernel(
    const unsigned short* __restrict__ Q,   // S x (H*D)
    const unsigned short* __restrict__ Km,  // S x (KVH*D)
    const unsigned short* __restrict__ Vm,  // S x (KVH*D)
    unsigned short* __restrict__ O)         // S x (H*D)
{
    __shared__ unsigned short sK[64][136];  // [key][dim], 272B rows
    __shared__ unsigned short sV[128][72];  // [dim][key] (transposed), 144B rows
    __shared__ unsigned short sP[128][72];  // staged P, per-wave-private 16-row strips

    const int tid  = threadIdx.x;
    const int lane = tid & 31;
    const int w    = tid >> 5;
    const int half = lane >> 4;
    const int ln   = lane & 15;
    const int h    = blockIdx.y;
    const int kvh  = h >> 2;                 // H/KVH = 4
    const int q0   = blockIdx.x * 128;
    const int qr0  = q0 + w * 16;
    const float scale = 0.08838834764831845f; // 1/sqrt(128)

    // Q fragments: 16 rows x 128 dims -> 4 K-steps of 32
    v16bf aq[4];
    {
        const unsigned short* qrow = Q + (size_t)(qr0 + ln) * (NH * HD) + h * HD;
#pragma unroll
        for (int ks = 0; ks < 4; ++ks) aq[ks] = load_fragA(qrow + ks * 32, half);
    }

    v8f zero = {0.f,0.f,0.f,0.f,0.f,0.f,0.f,0.f};
    v8f acc[8];
#pragma unroll
    for (int t = 0; t < 8; ++t) acc[t] = zero;
    float m_i[8], l_i[8];
#pragma unroll
    for (int i = 0; i < 8; ++i) { m_i[i] = -1e30f; l_i[i] = 0.0f; }

    // staging coords: thread handles one key row, 32 dims
    const int skey = tid >> 2;
    const int sd0  = (tid & 3) * 32;

    const int kc_end = q0 + 128;            // causal: keys beyond tile are masked
    for (int kc = 0; kc < kc_end; kc += 64) {
        {   // stage K chunk [key][dim]; V transposed [dim][key]
            const unsigned short* ksrc = Km + (size_t)(kc + skey) * (NKVH * HD) + kvh * HD + sd0;
            const unsigned short* vsrc = Vm + (size_t)(kc + skey) * (NKVH * HD) + kvh * HD + sd0;
            if (kc + 64 < kc_end) {
                __builtin_prefetch(ksrc + (size_t)64 * NKVH * HD, 0, 1);
                __builtin_prefetch(vsrc + (size_t)64 * NKVH * HD, 0, 1);
            }
            unsigned short* kdst = &sK[skey][sd0];
#if USE_ASYNC_LDS
#pragma unroll
            for (int j = 0; j < 4; ++j)
                __builtin_amdgcn_global_load_async_to_lds_b128(
                    (as1_v4i_p)(ksrc + j * 8), (as3_v4i_p)(kdst + j * 8), 0, 0);
#else
            const uint4* ks4 = (const uint4*)ksrc;
            uint4*       kd4 = (uint4*)kdst;
#pragma unroll
            for (int j = 0; j < 4; ++j) kd4[j] = ks4[j];
#endif
            const uint4* vs4 = (const uint4*)vsrc;
#pragma unroll
            for (int j = 0; j < 4; ++j) {
                uint4 t = vs4[j];
                unsigned uu[4] = {t.x, t.y, t.z, t.w};
#pragma unroll
                for (int e = 0; e < 4; ++e) {
                    sV[sd0 + j * 8 + 2 * e][skey]     = (unsigned short)(uu[e] & 0xffffu);
                    sV[sd0 + j * 8 + 2 * e + 1][skey] = (unsigned short)(uu[e] >> 16);
                }
            }
        }
#if USE_ASYNC_LDS
        __builtin_amdgcn_s_wait_asynccnt(0);
#endif
        __syncthreads();

        // S = Q * K^T : 16 x 64 per wave
        v8f sacc[4];
#pragma unroll
        for (int t = 0; t < 4; ++t) sacc[t] = zero;
#pragma unroll
        for (int ks = 0; ks < 4; ++ks) {
#pragma unroll
            for (int tn = 0; tn < 4; ++tn) {
                v16bf b = load_fragB(&sK[tn * 16 + ln][ks * 32], half);
                sacc[tn] = __builtin_amdgcn_wmma_f32_16x16x32_bf16(
                    false, aq[ks], false, b, (short)0, sacc[tn], false, false);
            }
        }

        // online softmax
#pragma unroll
        for (int i = 0; i < 8; ++i) {
            int row = qr0 + half * 8 + i;
            float mx = -1e30f;
#pragma unroll
            for (int tn = 0; tn < 4; ++tn) {
                int col = kc + tn * 16 + ln;
                float v = sacc[tn][i] * scale;
                if (col > row) v = -1e30f;
                sacc[tn][i] = v;
                mx = fmaxf(mx, v);
            }
#pragma unroll
            for (int msk = 1; msk < 16; msk <<= 1)     // reduce over N lanes
                mx = fmaxf(mx, __shfl_xor(mx, msk, 32));

            float mnew = fmaxf(m_i[i], mx);
            float corr = __expf(m_i[i] - mnew);
            float rs = 0.0f;
#pragma unroll
            for (int tn = 0; tn < 4; ++tn) {
                float p = __expf(sacc[tn][i] - mnew);
                sacc[tn][i] = p;
                rs += p;
            }
#pragma unroll
            for (int msk = 1; msk < 16; msk <<= 1)
                rs += __shfl_xor(rs, msk, 32);

            l_i[i] = l_i[i] * corr + rs;
            m_i[i] = mnew;
#pragma unroll
            for (int t = 0; t < 8; ++t) acc[t][i] *= corr;
            // restage P (C-layout -> LDS) for use as an A operand
#pragma unroll
            for (int tn = 0; tn < 4; ++tn)
                sP[w * 16 + half * 8 + i][tn * 16 + ln] = f2bf(sacc[tn][i]);
        }

        // O += P * V  (16x128 per wave; LDS in-order per wave covers sP RAW)
#pragma unroll
        for (int kp = 0; kp < 2; ++kp) {
            v16bf ap = load_fragA(&sP[w * 16 + ln][kp * 32], half);
#pragma unroll
            for (int t = 0; t < 8; ++t) {
                v16bf bv = load_fragB(&sV[t * 16 + ln][kp * 32], half);
                acc[t] = __builtin_amdgcn_wmma_f32_16x16x32_bf16(
                    false, ap, false, bv, (short)0, acc[t], false, false);
            }
        }
        __syncthreads();
    }

    // normalize + store (transpose(0,2,1,3) layout: row s, col h*D+d)
#pragma unroll
    for (int i = 0; i < 8; ++i) {
        float inv = 1.0f / l_i[i];
        int row = qr0 + half * 8 + i;
#pragma unroll
        for (int t = 0; t < 8; ++t)
            O[(size_t)row * (NH * HD) + h * HD + t * 16 + ln] = f2bf(acc[t][i] * inv);
    }
}

// ---------------------------------------------------------------------------
// Host launcher
// ---------------------------------------------------------------------------
extern "C" void kernel_launch(void* const* d_in, const int* in_sizes, int n_in,
                              void* d_out, int out_size, void* d_ws, size_t ws_size,
                              hipStream_t stream)
{
    (void)in_sizes; (void)n_in; (void)out_size; (void)ws_size;

    const float* hidden = (const float*)d_in[0];   // S x HID
    const float* wq     = (const float*)d_in[1];   // HID x (H*D)
    const float* wk     = (const float*)d_in[2];   // HID x (KVH*D)
    const float* wv     = (const float*)d_in[3];   // HID x (KVH*D)
    const float* wo     = (const float*)d_in[4];   // (H*D) x HID
    float*       out    = (float*)d_out;           // S x HID

    // workspace layout (bf16): q | k | v | attn  (~40 MB total)
    unsigned short* qbf    = (unsigned short*)d_ws;
    unsigned short* kbf    = qbf + (size_t)S_LEN * NH * HD;
    unsigned short* vbf    = kbf + (size_t)S_LEN * NKVH * HD;
    unsigned short* attnbf = vbf + (size_t)S_LEN * NKVH * HD;

    dim3 blk(256);

    // QKV projections
    gemm_wmma_kernel<false, true><<<dim3((NH * HD) / BN, S_LEN / BM), blk, 0, stream>>>(
        hidden, wq, qbf, S_LEN, NH * HD, HID);
    gemm_wmma_kernel<false, true><<<dim3((NKVH * HD) / BN, S_LEN / BM), blk, 0, stream>>>(
        hidden, wk, kbf, S_LEN, NKVH * HD, HID);
    gemm_wmma_kernel<false, true><<<dim3((NKVH * HD) / BN, S_LEN / BM), blk, 0, stream>>>(
        hidden, wv, vbf, S_LEN, NKVH * HD, HID);

    // RoPE on q and k
    {
        int qtot = S_LEN * NH * (HD / 2);
        int ktot = S_LEN * NKVH * (HD / 2);
        rope_kernel<<<dim3((qtot + 255) / 256), blk, 0, stream>>>(qbf, NH);
        rope_kernel<<<dim3((ktot + 255) / 256), blk, 0, stream>>>(kbf, NKVH);
    }

    // causal flash attention
    flash_attn_kernel<<<dim3(S_LEN / 128, NH), blk, 0, stream>>>(qbf, kbf, vbf, attnbf);

    // output projection (bf16 A, f32 out)
    gemm_wmma_kernel<true, false><<<dim3(HID / BN, S_LEN / BM), blk, 0, stream>>>(
        attnbf, wo, out, S_LEN, HID, NH * HD);
}
